// HeteroGNN_14164802142730
// MI455X (gfx1250) — matrix operand order repaired and checked by reading, not compiled, not run on hardware
//
#include <hip/hip_runtime.h>
#include <hip/hip_bf16.h>

typedef __attribute__((ext_vector_type(16))) _Float16 v16h;
typedef __attribute__((ext_vector_type(4)))  _Float16 v4h;
typedef __attribute__((ext_vector_type(8)))  float    v8f;

#define HDIM 64
#define APAD 72          // LDS row stride (halves) for A panel: 64 + 8
#define WPADD 8          // extra halves per Wt row in LDS

// ---------------------------------------------------------------------------
// Elementwise / utility kernels (float4-vectorized streaming)
// ---------------------------------------------------------------------------
__global__ void fill4_kernel(float4* __restrict__ p, int n4, float v) {
    int i = blockIdx.x * blockDim.x + threadIdx.x;
    if (i < n4) p[i] = make_float4(v, v, v, v);
}

__global__ void embed_kernel(const float4* __restrict__ emb, const int* __restrict__ feat,
                             float4* __restrict__ x, int n) {
    int i = blockIdx.x * blockDim.x + threadIdx.x;   // over n*16 float4s
    if (i < n * 16) {
        int node = i >> 4;
        x[i] = emb[(size_t)feat[node] * 16 + (i & 15)];
    }
}

// Transposing weight prep: dst is [64][Ktot] f16 (n-major), src is [64][64] f32
// (k-major, out = x @ W => W[k][n]).  dst[n*Ktot + koff + k] = s0[k*64+n] (+s1)
__global__ void convert_w_kernel(const float* __restrict__ s0, const float* __restrict__ s1,
                                 _Float16* __restrict__ dst, int koff, int Ktot) {
    int i = blockIdx.x * blockDim.x + threadIdx.x;
    if (i < HDIM * HDIM) {
        int k = i >> 6, n = i & 63;
        float v = s0[k * HDIM + n];
        if (s1) v += s1[k * HDIM + n];
        dst[n * Ktot + koff + k] = (_Float16)v;
    }
}

__global__ void combine_bias_kernel(const float* __restrict__ b0, const float* __restrict__ b1,
                                    float* __restrict__ out) {
    int c = threadIdx.x;
    out[c] = b0[c] + b1[c];
}

// 64 lanes per edge: coalesced gather + f32 atomic scatter through L2
__global__ void scatter_kernel(const float* __restrict__ xsrc, const int* __restrict__ src,
                               const int* __restrict__ dst, int nedges,
                               float* __restrict__ agg, float* __restrict__ cnt) {
    int t = blockIdx.x * blockDim.x + threadIdx.x;
    int e = t >> 6;
    int c = t & 63;
    if (e >= nedges) return;
    int s = src[e], d = dst[e];
    atomicAdd(&agg[d * HDIM + c], xsrc[s * HDIM + c]);
    if (c == 0) atomicAdd(&cnt[d], 1.0f);
}

__global__ void normalize_kernel(float4* __restrict__ agg, const float* __restrict__ cnt, int n) {
    int i = blockIdx.x * blockDim.x + threadIdx.x;   // over n*16 float4s
    if (i < n * 16) {
        float rc = 1.0f / fmaxf(cnt[i >> 4], 1.0f);
        float4 v = agg[i];
        v.x *= rc; v.y *= rc; v.z *= rc; v.w *= rc;
        agg[i] = v;
    }
}

// ---------------------------------------------------------------------------
// WMMA GEMM: out[n,64] = [A0|A1|A2][n, 64*nA] @ W[64*nA, 64] + bias (opt leaky)
// W is pre-transposed f16: W[n][k], densely packed [64][Ktot].
// Block = 256 threads = 8 wave32; wave -> 16x64 tile; block -> 128 rows.
// ---------------------------------------------------------------------------
__global__ __launch_bounds__(256)
void gemm_wmma_kernel(const float* __restrict__ A0, const float* __restrict__ A1,
                      const float* __restrict__ A2, int nA,
                      const _Float16* __restrict__ W, const float* __restrict__ bias,
                      float* __restrict__ out, int nrows, int act) {
    __shared__ _Float16 sW[HDIM * (192 + WPADD)];  // n-major, padded row stride
    __shared__ _Float16 sA[128 * APAD];            // row-major, padded row stride

    const int tid  = threadIdx.x;
    const int lane = tid & 31;
    const int wave = tid >> 5;
    const int rowBase = blockIdx.x * 128;
    const int col0 = lane & 15;

    const int Ktot  = nA * HDIM;
    const int KtotP = Ktot + WPADD;

    // stage transposed weights into LDS with padded stride (conflict-free B reads)
    for (int i = tid; i < HDIM * Ktot; i += 256) {
        int n = i / Ktot, k = i - n * Ktot;
        sW[n * KtotP + k] = W[i];
    }

    // accumulators seeded with bias (C layout: col fixed per lane = 16t+col0)
    v8f acc[4];
#pragma unroll
    for (int t = 0; t < 4; ++t) {
        float bv = bias[t * 16 + col0];
#pragma unroll
        for (int r = 0; r < 8; ++r) acc[t][r] = bv;
    }

    const float* As[3] = {A0, A1, A2};

    for (int s = 0; s < nA; ++s) {
        __syncthreads();
        const float4* Ap4 = (const float4*)As[s];
        // vectorized staging: global b128 load -> cvt -> ds b64 store
#pragma unroll
        for (int it = 0; it < 8; ++it) {
            int j = tid + it * 256;            // over 128*16 float4 chunks
            int r = j >> 4, c4 = j & 15;
            int gr = rowBase + r;
            float4 v = make_float4(0.f, 0.f, 0.f, 0.f);
            if (gr < nrows) v = Ap4[(size_t)gr * 16 + c4];
            v4h h;
            h[0] = (_Float16)v.x; h[1] = (_Float16)v.y;
            h[2] = (_Float16)v.z; h[3] = (_Float16)v.w;
            *(v4h*)&sA[r * APAD + (c4 << 2)] = h;
        }
        __syncthreads();

#pragma unroll
        for (int kt = 0; kt < 2; ++kt) {
            const int kb = kt * 32;
            // A fragment: 16x32 f16; lane holds M=lane&15;
            // halves: [base0, base0+16) contiguous; lanes 16-31 hold K+16..K+31
            const int m = (wave << 4) + (lane & 15);
            const int base0 = kb + ((lane >> 4) << 4);
            const _Float16* ap = &sA[m * APAD + base0];
            v16h a;
#pragma unroll
            for (int e = 0; e < 8; ++e) { a[e] = ap[e]; a[e + 8] = ap[e + 8]; }

            const int kg  = s * HDIM + kb;
            const int kbB = kg + ((lane >> 4) << 4);   // lanes 16-31 hold K+16..K+31
#pragma unroll
            for (int t = 0; t < 4; ++t) {
                const int n = t * 16 + col0;
                const _Float16* bp = &sW[n * KtotP + kbB];  // 16 contiguous halves
                v16h b;
#pragma unroll
                for (int e = 0; e < 16; ++e) b[e] = bp[e];
                acc[t] = __builtin_amdgcn_wmma_f32_16x16x32_f16(
                    false, a, false, b, (short)0, acc[t], false, false);
            }
        }
    }

    // C/D layout: VGPR r -> row r (lanes 0-15) / r+8 (lanes 16-31)
    const int rhalf = (lane >> 4) << 3;
#pragma unroll
    for (int t = 0; t < 4; ++t) {
        const int col = t * 16 + col0;
#pragma unroll
        for (int r = 0; r < 8; ++r) {
            int row = rowBase + (wave << 4) + rhalf + r;
            if (row < nrows) {
                float v = acc[t][r];
                if (act) v = (v >= 0.0f) ? v : 0.01f * v;
                out[(size_t)row * HDIM + col] = v;
            }
        }
    }
}

// ---------------------------------------------------------------------------
// BatchNorm (training mode, biased variance) over node dim
// ---------------------------------------------------------------------------
__global__ void bn_stats_kernel(const float* __restrict__ x, int n, float* __restrict__ stats) {
    __shared__ float s1[256], s2[256];
    int tid = threadIdx.x;
    int c = tid & 63;
    float a = 0.0f, b = 0.0f;
    for (int r = blockIdx.x * 4 + (tid >> 6); r < n; r += gridDim.x * 4) {
        float v = x[(size_t)r * HDIM + c];
        a += v;
        b += v * v;
    }
    s1[tid] = a; s2[tid] = b;
    __syncthreads();
    if (tid < 64) {
        a = s1[tid] + s1[tid + 64] + s1[tid + 128] + s1[tid + 192];
        b = s2[tid] + s2[tid + 64] + s2[tid + 128] + s2[tid + 192];
        atomicAdd(&stats[tid], a);
        atomicAdd(&stats[64 + tid], b);
    }
}

__global__ void bn_finalize_kernel(float* __restrict__ stats, int n,
                                   const float* __restrict__ gamma,
                                   const float* __restrict__ beta) {
    int c = threadIdx.x;  // 64
    float inv_n = 1.0f / (float)n;
    float mu = stats[c] * inv_n;
    float var = stats[64 + c] * inv_n - mu * mu;
    float sc = gamma[c] * rsqrtf(var + 1e-5f);
    stats[c] = sc;
    stats[64 + c] = beta[c] - mu * sc;
}

__global__ void bn_apply_kernel(const float4* __restrict__ x, const float* __restrict__ stats,
                                float4* __restrict__ out, int n, int act) {
    int i = blockIdx.x * blockDim.x + threadIdx.x;   // over n*16 float4s
    if (i < n * 16) {
        int c4 = (i & 15) << 2;
        float4 v = x[i];
        float4 r;
        r.x = stats[c4 + 0] * v.x + stats[64 + c4 + 0];
        r.y = stats[c4 + 1] * v.y + stats[64 + c4 + 1];
        r.z = stats[c4 + 2] * v.z + stats[64 + c4 + 2];
        r.w = stats[c4 + 3] * v.w + stats[64 + c4 + 3];
        if (act) {
            r.x = (r.x >= 0.0f) ? r.x : 0.01f * r.x;
            r.y = (r.y >= 0.0f) ? r.y : 0.01f * r.y;
            r.z = (r.z >= 0.0f) ? r.z : 0.01f * r.z;
            r.w = (r.w >= 0.0f) ? r.w : 0.01f * r.w;
        }
        out[i] = r;
    }
}

// ---------------------------------------------------------------------------
// Edge scoring: 32 lanes per edge, wave32 shuffle reduction
// ---------------------------------------------------------------------------
__global__ void score_kernel(const float* __restrict__ xa, const float* __restrict__ xb,
                             const int* __restrict__ e0, const int* __restrict__ e1,
                             const float* __restrict__ lab_src,
                             float* __restrict__ pred_out, float* __restrict__ lab_out, int nE) {
    int w = (blockIdx.x * blockDim.x + threadIdx.x) >> 5;
    int lane = threadIdx.x & 31;
    if (w >= nE) return;
    int ia = e0[w], ib = e1[w];
    float v = xa[(size_t)ia * HDIM + lane] * xb[(size_t)ib * HDIM + lane]
            + xa[(size_t)ia * HDIM + 32 + lane] * xb[(size_t)ib * HDIM + 32 + lane];
#pragma unroll
    for (int off = 16; off > 0; off >>= 1) v += __shfl_xor(v, off, 32);
    if (lane == 0) {
        pred_out[w] = v;
        lab_out[w]  = lab_src[ib];
    }
}

// ---------------------------------------------------------------------------
extern "C" void kernel_launch(void* const* d_in, const int* in_sizes, int n_in,
                              void* d_out, int out_size, void* d_ws, size_t ws_size,
                              hipStream_t stream) {
    const float* emb_obj    = (const float*)d_in[0];
    const float* emb_attr   = (const float*)d_in[1];
    const float* mlp_w_obj  = (const float*)d_in[2];
    const float* mlp_b_obj  = (const float*)d_in[3];
    const float* mlp_w_attr = (const float*)d_in[4];
    const float* mlp_b_attr = (const float*)d_in[5];
    const float* conv_wself = (const float*)d_in[6];
    const float* conv_wmsg  = (const float*)d_in[7];
    const float* conv_b     = (const float*)d_in[8];
    const float* bn_gamma   = (const float*)d_in[9];
    const float* bn_beta    = (const float*)d_in[10];
    const float* lab_obj    = (const float*)d_in[11];
    const float* lab_attr   = (const float*)d_in[12];
    const int* feat_obj = (const int*)d_in[13];
    const int* feat_attr = (const int*)d_in[14];
    const int* ei0 = (const int*)d_in[15];
    const int* ei1 = (const int*)d_in[16];
    const int* ei2 = (const int*)d_in[17];
    const int* eli0 = (const int*)d_in[18];
    const int* eli1 = (const int*)d_in[19];
    const int* eli2 = (const int*)d_in[20];

    const int N  = in_sizes[13];
    const int E  = in_sizes[15] / 2;
    const int EL = in_sizes[18] / 2;
    float* out = (float*)d_out;

    // workspace layout
    float* ws = (float*)d_ws;
    size_t NH = (size_t)N * HDIM;
    float* x_obj    = ws;
    float* x_attr   = x_obj + NH;
    float* tmp      = x_attr + NH;      // MLP ping / new_obj
    float* new_attr = tmp + NH;
    float* aggA     = new_attr + NH;
    float* aggB     = aggA + NH;
    float* cntA     = aggB + NH;
    float* cntB     = cntA + N;
    float* bcomb    = cntB + N;
    float* stats    = bcomb + HDIM;
    _Float16* Wc    = (_Float16*)(stats + 128);

    dim3 blk(256);
    int gVec   = (N * 16 + 255) / 256;          // float4 sweeps over [N,64]
    int gNode4 = (N / 4 + 255) / 256;           // float4 fill over [N]
    int gGemm  = (N + 127) / 128;
    int gEdge  = (int)(((long long)E * HDIM + 255) / 256);
    int gW     = (4096 + 255) / 256;
    int gScore = (EL * 32 + 255) / 256;

    // --- embedding lookup ---
    embed_kernel<<<gVec, blk, 0, stream>>>((const float4*)emb_obj, feat_obj, (float4*)x_obj, N);
    embed_kernel<<<gVec, blk, 0, stream>>>((const float4*)emb_attr, feat_attr, (float4*)x_attr, N);

    // --- encoder MLPs (2 layers, Linear + LeakyReLU) ---
    for (int i = 0; i < 2; ++i) {
        convert_w_kernel<<<gW, blk, 0, stream>>>(mlp_w_obj + i * 4096, nullptr, Wc, 0, 64);
        gemm_wmma_kernel<<<gGemm, blk, 0, stream>>>(
            i == 0 ? x_obj : tmp, x_obj, x_obj, 1, Wc, mlp_b_obj + i * HDIM,
            i == 0 ? tmp : x_obj, N, 1);
    }
    for (int i = 0; i < 2; ++i) {
        convert_w_kernel<<<gW, blk, 0, stream>>>(mlp_w_attr + i * 4096, nullptr, Wc, 0, 64);
        gemm_wmma_kernel<<<gGemm, blk, 0, stream>>>(
            i == 0 ? x_attr : new_attr, x_attr, x_attr, 1, Wc, mlp_b_attr + i * HDIM,
            i == 0 ? new_attr : x_attr, N, 1);
    }

    // --- hetero SAGE layers ---
    for (int l = 0; l < 2; ++l) {
        // message type 0: obj -> attr  (K=128 fused GEMM)
        fill4_kernel<<<gVec, blk, 0, stream>>>((float4*)aggA, N * 16, 0.0f);
        fill4_kernel<<<gNode4, blk, 0, stream>>>((float4*)cntA, N / 4, 0.0f);
        scatter_kernel<<<gEdge, blk, 0, stream>>>(x_obj, ei0, ei0 + E, E, aggA, cntA);
        normalize_kernel<<<gVec, blk, 0, stream>>>((float4*)aggA, cntA, N);
        convert_w_kernel<<<gW, blk, 0, stream>>>(conv_wself + (l * 3 + 0) * 4096, nullptr, Wc, 0, 128);
        convert_w_kernel<<<gW, blk, 0, stream>>>(conv_wmsg + (l * 3 + 0) * 4096, nullptr, Wc, 64, 128);
        gemm_wmma_kernel<<<gGemm, blk, 0, stream>>>(
            x_attr, aggA, aggA, 2, Wc, conv_b + (l * 3 + 0) * HDIM, new_attr, N, 0);

        // message types 1 (attr->obj) + 2 (obj->obj), fused K=192 GEMM
        fill4_kernel<<<gVec, blk, 0, stream>>>((float4*)aggA, N * 16, 0.0f);
        fill4_kernel<<<gVec, blk, 0, stream>>>((float4*)aggB, N * 16, 0.0f);
        fill4_kernel<<<gNode4, blk, 0, stream>>>((float4*)cntA, N / 4, 0.0f);
        fill4_kernel<<<gNode4, blk, 0, stream>>>((float4*)cntB, N / 4, 0.0f);
        scatter_kernel<<<gEdge, blk, 0, stream>>>(x_attr, ei1, ei1 + E, E, aggA, cntA);
        scatter_kernel<<<gEdge, blk, 0, stream>>>(x_obj, ei2, ei2 + E, E, aggB, cntB);
        normalize_kernel<<<gVec, blk, 0, stream>>>((float4*)aggA, cntA, N);
        normalize_kernel<<<gVec, blk, 0, stream>>>((float4*)aggB, cntB, N);
        convert_w_kernel<<<gW, blk, 0, stream>>>(conv_wself + (l * 3 + 1) * 4096,
                                                 conv_wself + (l * 3 + 2) * 4096, Wc, 0, 192);
        convert_w_kernel<<<gW, blk, 0, stream>>>(conv_wmsg + (l * 3 + 1) * 4096, nullptr, Wc, 64, 192);
        convert_w_kernel<<<gW, blk, 0, stream>>>(conv_wmsg + (l * 3 + 2) * 4096, nullptr, Wc, 128, 192);
        combine_bias_kernel<<<1, HDIM, 0, stream>>>(conv_b + (l * 3 + 1) * HDIM,
                                                    conv_b + (l * 3 + 2) * HDIM, bcomb);
        gemm_wmma_kernel<<<gGemm, blk, 0, stream>>>(x_obj, aggA, aggB, 3, Wc, bcomb, tmp, N, 0);

        int act = (l == 0) ? 1 : 0;
        // BN obj
        fill4_kernel<<<1, 32, 0, stream>>>((float4*)stats, 32, 0.0f);
        bn_stats_kernel<<<512, blk, 0, stream>>>(tmp, N, stats);
        bn_finalize_kernel<<<1, HDIM, 0, stream>>>(stats, N,
            bn_gamma + (l * 2 + 0) * HDIM, bn_beta + (l * 2 + 0) * HDIM);
        bn_apply_kernel<<<gVec, blk, 0, stream>>>((const float4*)tmp, stats, (float4*)x_obj, N, act);
        // BN attr
        fill4_kernel<<<1, 32, 0, stream>>>((float4*)stats, 32, 0.0f);
        bn_stats_kernel<<<512, blk, 0, stream>>>(new_attr, N, stats);
        bn_finalize_kernel<<<1, HDIM, 0, stream>>>(stats, N,
            bn_gamma + (l * 2 + 1) * HDIM, bn_beta + (l * 2 + 1) * HDIM);
        bn_apply_kernel<<<gVec, blk, 0, stream>>>((const float4*)new_attr, stats, (float4*)x_attr, N, act);
    }

    // --- edge scoring + label gather ---
    score_kernel<<<gScore, blk, 0, stream>>>(x_obj, x_attr, eli0, eli0 + EL, lab_attr,
                                             out + 0 * EL, out + 3 * EL, EL);
    score_kernel<<<gScore, blk, 0, stream>>>(x_attr, x_obj, eli1, eli1 + EL, lab_obj,
                                             out + 1 * EL, out + 4 * EL, EL);
    score_kernel<<<gScore, blk, 0, stream>>>(x_obj, x_obj, eli2, eli2 + EL, lab_obj,
                                             out + 2 * EL, out + 5 * EL, EL);

    (void)n_in; (void)out_size; (void)ws_size;
}
